// unite_25039659336156
// MI455X (gfx1250) — matrix operand order repaired and checked by reading, not compiled, not run on hardware
//
#include <hip/hip_runtime.h>
#include <math.h>

// ---------------------------------------------------------------------------
// Problem constants
// ---------------------------------------------------------------------------
#define BB 64
#define TT 365
#define NN (BB * TT)          // 23360 rows, exactly 1460 tiles of 16
#define DMODEL 256

typedef _Float16 v16h __attribute__((ext_vector_type(16)));
typedef _Float16 h8   __attribute__((ext_vector_type(8)));
typedef float    v8f  __attribute__((ext_vector_type(8)));

// ---------------------------------------------------------------------------
// Utility: zero a float region (stats accumulators) -- must run every call
// ---------------------------------------------------------------------------
__global__ void zero_f32_k(float* p, int n) {
    int i = blockIdx.x * blockDim.x + threadIdx.x;
    if (i < n) p[i] = 0.0f;
}

// ---------------------------------------------------------------------------
// Weight conversion: fp32 [NOUT,KDIM] -> f16 [NOUT,KPAD] zero-padded rows.
// Runs once per call; makes GEMM B-fragment loads aligned 16-byte vectors.
// ---------------------------------------------------------------------------
__global__ void wconv_k(const float* __restrict__ W, _Float16* __restrict__ Wf,
                        int NOUT, int KDIM, int KPAD) {
    int i = blockIdx.x * blockDim.x + threadIdx.x;
    if (i < NOUT * KPAD) {
        int n = i / KPAD, k = i % KPAD;
        Wf[i] = (k < KDIM) ? (_Float16)W[n * KDIM + k] : (_Float16)0.0f;
    }
}

// ---------------------------------------------------------------------------
// WMMA GEMM + BN-stat accumulation.
//   Y[N, NOUT] = transform(A)[N, KDIM] @ W[NOUT, KDIM]^T + bias
//   TRANSFORM: 0 = A is raw fp32 (layer 1, KDIM=10 zero-padded)
//              1 = A is f16 pre-BN linear output; apply max(0, a*scale+shift)
//              2 = A is f16, use as-is
//   Wf is the pre-converted f16 weight with row stride KPAD.
//   Per-feature sum/sumsq: lane row-sums -> LDS atomics -> 1 global atomic/feat.
//   Output: D tiles scattered into LDS, then streamed out as coalesced b128.
//   Grid: NN/16 blocks of 256 threads (8 waves), wave = 16x16 output tiles.
// ---------------------------------------------------------------------------
template <int KDIM, int NOUT, int TRANSFORM>
__global__ void gemm_bn_k(const void* __restrict__ Ain,
                          const _Float16* __restrict__ Wf,
                          const float* __restrict__ bias,
                          const float* __restrict__ in_scale,
                          const float* __restrict__ in_shift,
                          _Float16* __restrict__ Yout,
                          float* __restrict__ stat_sum,
                          float* __restrict__ stat_sq) {
    constexpr int KPAD   = ((KDIM + 31) / 32) * 32;
    constexpr int KROW   = KPAD + 8;              // 16B-aligned row stride, bank spread
    constexpr int NCHUNK = KPAD / 32;
    __shared__ __align__(16) _Float16 As[16][KROW];
    __shared__ __align__(16) _Float16 Ds[16][NOUT];
    __shared__ float redsum[NOUT];
    __shared__ float redsq[NOUT];

    const int tid  = threadIdx.x;
    const int row0 = blockIdx.x * 16;

    // ---- Stage 16 x KPAD input tile into LDS (transform applied in regs) ----
    if (TRANSFORM == 0) {
        // fp32 input, KDIM not a multiple of 8: scalar path with zero pad.
        for (int idx = tid; idx < 16 * KPAD; idx += 256) {
            int r = idx / KPAD, k = idx % KPAD;
            float v = (k < KDIM)
                ? ((const float*)Ain)[(size_t)(row0 + r) * KDIM + k] : 0.0f;
            As[r][k] = (_Float16)v;
        }
    } else {
        // f16 input, KDIM == KPAD multiple of 8: 16-byte vector path.
        constexpr int KV = KDIM / 8;
        for (int u = tid; u < 16 * KV; u += 256) {
            int r = u / KV, ko = (u % KV) * 8;
            h8 raw = *(const h8*)((const _Float16*)Ain +
                                  (size_t)(row0 + r) * KDIM + ko);
            h8 o;
#pragma unroll
            for (int j = 0; j < 8; ++j) {
                float v = (float)raw[j];
                if (TRANSFORM == 1)
                    v = fmaxf(0.0f, v * in_scale[ko + j] + in_shift[ko + j]);
                o[j] = (_Float16)v;
            }
            *(h8*)&As[r][ko] = o;
        }
    }
    for (int j = tid; j < NOUT; j += 256) { redsum[j] = 0.0f; redsq[j] = 0.0f; }
    __syncthreads();

    const int wid  = tid >> 5;
    const int lane = tid & 31;
    const int l    = lane & 15;
    const int hi   = lane >> 4;

    // ---- Hoist all A fragments into registers (shared across col tiles) ----
    // 16-bit A 16x32 layout: lane hi-half holds K {0-7,16-23} vs {8-15,24-31}.
    v16h af[NCHUNK];
#pragma unroll
    for (int c = 0; c < NCHUNK; ++c) {
        h8 lo8 = *(const h8*)&As[l][c * 32 + hi * 8];
        h8 hi8 = *(const h8*)&As[l][c * 32 + 16 + hi * 8];
#pragma unroll
        for (int j = 0; j < 8; ++j) { af[c][j] = lo8[j]; af[c][8 + j] = hi8[j]; }
    }

    for (int ct = wid; ct < NOUT / 16; ct += 8) {
        const int col0 = ct * 16;
        // B fragment: B[k][n] = W[n][k]; lane hi-half holds K {0-7,16-23} vs {8-15,24-31}.
        const _Float16* wrow = Wf + (size_t)(col0 + l) * KPAD + hi * 8;
        v8f acc = {};
#pragma unroll
        for (int c = 0; c < NCHUNK; ++c) {
            h8 b0 = *(const h8*)(wrow + c * 32);
            h8 b1 = *(const h8*)(wrow + c * 32 + 16);
            v16h b;
#pragma unroll
            for (int j = 0; j < 8; ++j) { b[j] = b0[j]; b[8 + j] = b1[j]; }
            acc = __builtin_amdgcn_wmma_f32_16x16x32_f16(
                false, af[c], false, b, (short)0, acc, false, false);
        }
        // D layout: VGPR r -> row = hi*8 + r, col = col0 + l. Scatter to LDS;
        // per-feature partial stats from the accumulator (2 LDS atomics/lane).
        const float bn = bias[col0 + l];
        float s = 0.0f, sq = 0.0f;
#pragma unroll
        for (int r = 0; r < 8; ++r) {
            float v = acc[r] + bn;
            Ds[hi * 8 + r][col0 + l] = (_Float16)v;
            s += v; sq += v * v;
        }
        atomicAdd(&redsum[col0 + l], s);
        atomicAdd(&redsq[col0 + l], sq);
    }
    __syncthreads();

    // ---- Coalesced output: row-contiguous 16-byte stores ----
    constexpr int NV = NOUT / 8;
    for (int u = tid; u < 16 * NV; u += 256) {
        int r = u / NV, co = (u % NV) * 8;
        *(h8*)(Yout + (size_t)(row0 + r) * NOUT + co) = *(const h8*)&Ds[r][co];
    }
    // ---- One global atomic per feature per block ----
    for (int j = tid; j < NOUT; j += 256) {
        atomicAdd(&stat_sum[j], redsum[j]);
        atomicAdd(&stat_sq[j],  redsq[j]);
    }
}

// ---------------------------------------------------------------------------
// BN finalize: scale = g*rsqrt(var+eps), shift = beta - mu*scale
// ---------------------------------------------------------------------------
__global__ void bn_finalize_k(const float* __restrict__ sum, const float* __restrict__ sq,
                              const float* __restrict__ g, const float* __restrict__ beta,
                              float* __restrict__ scale, float* __restrict__ shift,
                              int nfeat, float invN) {
    int j = blockIdx.x * blockDim.x + threadIdx.x;
    if (j < nfeat) {
        float mu  = sum[j] * invN;
        float var = sq[j] * invN - mu * mu;
        float sc  = rsqrtf(var + 1e-5f) * g[j];
        scale[j] = sc;
        shift[j] = beta[j] - mu * sc;
    }
}

// ---------------------------------------------------------------------------
// SE block part 1: pooled over time + 2 tiny linears + sigmoid.
// sepc[b,c] = att[b,c] * pooled[b,c].  Grid: BB blocks x 256 threads.
// ---------------------------------------------------------------------------
__global__ void se_pool_k(const _Float16* __restrict__ y3,
                          const float* __restrict__ sc3, const float* __restrict__ sh3,
                          const float* __restrict__ ws1, const float* __restrict__ ws2,
                          float* __restrict__ sepc) {
    __shared__ float pl[256];
    __shared__ float tmp[16];
    const int b = blockIdx.x, c = threadIdx.x;
    float s = 0.0f;
    const float scc = sc3[c], shc = sh3[c];
    for (int t = 0; t < TT; ++t) {
        float raw = (float)y3[((size_t)(b * TT + t)) * 256 + c];
        s += fmaxf(0.0f, raw * scc + shc);
    }
    pl[c] = s * (1.0f / TT);
    __syncthreads();
    if (c < 16) {
        float a = 0.0f;
        const float* wr = ws1 + c * 256;
        for (int j = 0; j < 256; ++j) a += pl[j] * wr[j];
        tmp[c] = fmaxf(0.0f, a);
    }
    __syncthreads();
    float a = 0.0f;
    const float* wr2 = ws2 + c * 16;
#pragma unroll
    for (int j = 0; j < 16; ++j) a += tmp[j] * wr2[j];
    float att = 1.0f / (1.0f + __expf(-a));
    sepc[b * 256 + c] = att * pl[c];
}

// SE part 2: outp = normrelu3(y3) + sepc (broadcast over t). Grid NN x 256.
__global__ void se_apply_k(const _Float16* __restrict__ y3,
                           const float* __restrict__ sc3, const float* __restrict__ sh3,
                           const float* __restrict__ sepc,
                           _Float16* __restrict__ outp) {
    const int n = blockIdx.x, c = threadIdx.x, b = n / TT;
    float raw = (float)y3[(size_t)n * 256 + c];
    float o = fmaxf(0.0f, raw * sc3[c] + sh3[c]) + sepc[b * 256 + c];
    outp[(size_t)n * 256 + c] = (_Float16)o;
}

// ---------------------------------------------------------------------------
// e = normrelu4(y4) + sinusoidal positional encoding (computed on device).
// angle = pos / 1000^(2*(c/2)/256); even c -> sin, odd -> cos. Grid NN x 256.
// ---------------------------------------------------------------------------
__global__ void e_build_k(const _Float16* __restrict__ y4,
                          const float* __restrict__ sc4, const float* __restrict__ sh4,
                          const int* __restrict__ positions,
                          _Float16* __restrict__ ebuf) {
    const int n = blockIdx.x, c = threadIdx.x;
    float raw = (float)y4[(size_t)n * 256 + c];
    float h = fmaxf(0.0f, raw * sc4[c] + sh4[c]);
    float pos = (float)positions[n];
    float ex = (float)(2 * (c >> 1)) * (1.0f / 256.0f);
    float angle = pos * __expf(-ex * 6.9077552790f);   // 1000^-ex
    float pe = (c & 1) ? __cosf(angle) : __sinf(angle);
    ebuf[(size_t)n * 256 + c] = (_Float16)(h + pe);
}

// ---------------------------------------------------------------------------
// kbar = (mean_t e) @ wk.T + bk ; fold Q projection:
//   wvec[b,h,:] = (1/sqrt8) * sum_d kbar_d * wq[h*8+d,:]
//   cconst[b,h] = (1/sqrt8) * sum_d bq[h*8+d] * kbar_d
// Grid: BB blocks x 256 threads.
// ---------------------------------------------------------------------------
__global__ void kbar_wvec_k(const _Float16* __restrict__ ebuf,
                            const float* __restrict__ wk, const float* __restrict__ bk,
                            const float* __restrict__ wq, const float* __restrict__ bq,
                            float* __restrict__ wvec, float* __restrict__ cconst) {
    __shared__ float ebar[256];
    __shared__ float kb[128];
    const int b = blockIdx.x, tid = threadIdx.x;
    float s = 0.0f;
    for (int t = 0; t < TT; ++t) s += (float)ebuf[((size_t)(b * TT + t)) * 256 + tid];
    ebar[tid] = s * (1.0f / TT);
    __syncthreads();
    if (tid < 128) {
        float a = bk[tid];
        const float* wr = wk + tid * 256;
        for (int c = 0; c < 256; ++c) a += ebar[c] * wr[c];
        kb[tid] = a;
    }
    __syncthreads();
    const float inv = 0.35355339059327373f;  // 1/sqrt(8)
    for (int h = 0; h < 16; ++h) {
        float wv = 0.0f;
#pragma unroll
        for (int d = 0; d < 8; ++d) wv += kb[h * 8 + d] * wq[(size_t)(h * 8 + d) * 256 + tid];
        wvec[((size_t)(b * 16 + h)) * 256 + tid] = wv * inv;
    }
    if (tid < 16) {
        float cc = 0.0f;
#pragma unroll
        for (int d = 0; d < 8; ++d) cc += bq[tid * 8 + d] * kb[tid * 8 + d];
        cconst[b * 16 + tid] = cc * inv;
    }
}

// scores[b,h,t] = e[b,t,:]·wvec[b,h,:] + cconst[b,h].  One thread each;
// 16 consecutive threads share one e row (broadcast loads).
__global__ void scores_k(const _Float16* __restrict__ ebuf,
                         const float* __restrict__ wvec, const float* __restrict__ cconst,
                         float* __restrict__ scores) {
    int gid = blockIdx.x * blockDim.x + threadIdx.x;
    if (gid >= BB * TT * 16) return;
    int h = gid & 15, rest = gid >> 4;
    int t = rest % TT, b = rest / TT;
    const _Float16* er = ebuf + ((size_t)(b * TT + t)) * 256;
    const float* wv = wvec + ((size_t)(b * 16 + h)) * 256;
    float a = cconst[b * 16 + h];
    for (int c = 0; c < 256; ++c) a += (float)er[c] * wv[c];
    scores[((size_t)(b * 16 + h)) * TT + t] = a;
}

// ---------------------------------------------------------------------------
// Per-(b,h): softmax over t, then y[b, h*16+e] = sum_t p_t * e[b,t,h*16+e].
// Grid: BB*16 blocks x 128 threads.
// ---------------------------------------------------------------------------
__global__ void softmax_y_k(const float* __restrict__ scores,
                            const _Float16* __restrict__ ebuf,
                            float* __restrict__ yvec) {
    __shared__ float red[128];
    const int b = blockIdx.x >> 4, h = blockIdx.x & 15, tid = threadIdx.x;
    const float* sc = scores + ((size_t)(b * 16 + h)) * TT;

    float mx = -1e30f;
    for (int t = tid; t < TT; t += 128) mx = fmaxf(mx, sc[t]);
    red[tid] = mx; __syncthreads();
    for (int off = 64; off; off >>= 1) {
        if (tid < off) red[tid] = fmaxf(red[tid], red[tid + off]);
        __syncthreads();
    }
    mx = red[0]; __syncthreads();

    float z = 0.0f;
    for (int t = tid; t < TT; t += 128) z += __expf(sc[t] - mx);
    red[tid] = z; __syncthreads();
    for (int off = 64; off; off >>= 1) {
        if (tid < off) red[tid] += red[tid + off];
        __syncthreads();
    }
    float invZ = 1.0f / red[0]; __syncthreads();

    const int e = tid & 15, g = tid >> 4;  // 16 e x 8 t-groups
    float p = 0.0f;
    for (int t = g; t < TT; t += 8)
        p += __expf(sc[t] - mx) * (float)ebuf[((size_t)(b * TT + t)) * 256 + h * 16 + e];
    red[e * 8 + g] = p; __syncthreads();
    if (tid < 16) {
        float tot = 0.0f;
#pragma unroll
        for (int gg = 0; gg < 8; ++gg) tot += red[tid * 8 + gg];
        yvec[b * 256 + h * 16 + tid] = tot * invZ;
    }
}

// ---------------------------------------------------------------------------
// Fused head: y[64,256] -> (256->128 BN ReLU) -> (128->64) -> (64->32)
//             -> (32->10) logits. Single block, everything in LDS.
// ---------------------------------------------------------------------------
__device__ __forceinline__ void bn_block(float* s, int rows, int F,
                                         const float* g, const float* beta, int tid) {
    __shared__ float mus[128];
    __shared__ float rss[128];
    if (tid < F) {
        float m = 0.0f, q = 0.0f;
        for (int r = 0; r < rows; ++r) { float v = s[r * F + tid]; m += v; q += v * v; }
        float invR = 1.0f / (float)rows;
        m *= invR; q = q * invR - m * m;
        mus[tid] = m; rss[tid] = rsqrtf(q + 1e-5f);
    }
    __syncthreads();
    for (int idx = tid; idx < rows * F; idx += 256) {
        int j = idx % F;
        float v = (s[idx] - mus[j]) * rss[j] * g[j] + beta[j];
        s[idx] = fmaxf(v, 0.0f);
    }
    __syncthreads();
}

__global__ void head_mlp_k(const float* __restrict__ yv,
                           const float* wm, const float* bm, const float* gm, const float* em,
                           const float* wd1, const float* bd1, const float* gd1, const float* ed1,
                           const float* wd2, const float* bd2, const float* gd2, const float* ed2,
                           const float* wc, const float* bc,
                           float* __restrict__ out) {
    __shared__ float sA[64 * 128];
    __shared__ float sB[64 * 64];
    const int tid = threadIdx.x;
    // mlp: 256 -> 128
    for (int idx = tid; idx < 64 * 128; idx += 256) {
        int r = idx >> 7, j = idx & 127;
        float a = bm[j];
        const float* xr = yv + r * 256;
        const float* wr = wm + j * 256;
        for (int c = 0; c < 256; ++c) a += xr[c] * wr[c];
        sA[idx] = a;
    }
    __syncthreads();
    bn_block(sA, 64, 128, gm, em, tid);
    // decoder 1: 128 -> 64
    for (int idx = tid; idx < 64 * 64; idx += 256) {
        int r = idx >> 6, j = idx & 63;
        float a = bd1[j];
        for (int c = 0; c < 128; ++c) a += sA[r * 128 + c] * wd1[j * 128 + c];
        sB[idx] = a;
    }
    __syncthreads();
    bn_block(sB, 64, 64, gd1, ed1, tid);
    // decoder 2: 64 -> 32 (into sA, stage-1 data is dead)
    for (int idx = tid; idx < 64 * 32; idx += 256) {
        int r = idx >> 5, j = idx & 31;
        float a = bd2[j];
        for (int c = 0; c < 64; ++c) a += sB[r * 64 + c] * wd2[j * 64 + c];
        sA[idx] = a;
    }
    __syncthreads();
    bn_block(sA, 64, 32, gd2, ed2, tid);
    // classifier: 32 -> 10
    for (int idx = tid; idx < 64 * 10; idx += 256) {
        int r = idx / 10, j = idx % 10;
        float a = bc[j];
#pragma unroll
        for (int c = 0; c < 32; ++c) a += sA[r * 32 + c] * wc[j * 32 + c];
        out[idx] = a;
    }
}

// ---------------------------------------------------------------------------
// Launcher
// ---------------------------------------------------------------------------
extern "C" void kernel_launch(void* const* d_in, const int* in_sizes, int n_in,
                              void* d_out, int out_size, void* d_ws, size_t ws_size,
                              hipStream_t stream) {
    const float* x         = (const float*)d_in[0];
    const int*   positions = (const int*)d_in[1];
    const float *w1 = (const float*)d_in[2],  *b1 = (const float*)d_in[3],
                *g1 = (const float*)d_in[4],  *e1 = (const float*)d_in[5];
    const float *w2 = (const float*)d_in[6],  *b2 = (const float*)d_in[7],
                *g2 = (const float*)d_in[8],  *e2 = (const float*)d_in[9];
    const float *w3 = (const float*)d_in[10], *b3 = (const float*)d_in[11],
                *g3 = (const float*)d_in[12], *e3 = (const float*)d_in[13];
    const float *ws1 = (const float*)d_in[14], *ws2 = (const float*)d_in[15];
    const float *wi = (const float*)d_in[16], *bi = (const float*)d_in[17],
                *gi = (const float*)d_in[18], *ei = (const float*)d_in[19];
    const float *wk = (const float*)d_in[20], *bk = (const float*)d_in[21];
    const float *wq = (const float*)d_in[22], *bq = (const float*)d_in[23];
    const float *wm = (const float*)d_in[24], *bm = (const float*)d_in[25],
                *gm = (const float*)d_in[26], *em = (const float*)d_in[27];
    const float *wd1 = (const float*)d_in[28], *bd1 = (const float*)d_in[29],
                *gd1 = (const float*)d_in[30], *ed1 = (const float*)d_in[31];
    const float *wd2 = (const float*)d_in[32], *bd2 = (const float*)d_in[33],
                *gd2 = (const float*)d_in[34], *ed2 = (const float*)d_in[35];
    const float *wc = (const float*)d_in[36], *bc = (const float*)d_in[37];
    float* out = (float*)d_out;

    // Workspace layout: three 12 MB f16 rotating buffers + fp32 scratch.
    char* ws = (char*)d_ws;
    _Float16* bufA = (_Float16*)(ws);                            // y3, ebuf
    _Float16* bufB = (_Float16*)(ws + (size_t)12 * 1024 * 1024); // y2, outp
    _Float16* bufC = (_Float16*)(ws + (size_t)24 * 1024 * 1024); // y1, y4
    float* S = (float*)(ws + (size_t)36 * 1024 * 1024);
    float *sum1 = S + 0,    *sq1 = S + 64,   *sc1 = S + 128,  *sh1 = S + 192;
    float *sum2 = S + 256,  *sq2 = S + 384,  *sc2 = S + 512,  *sh2 = S + 640;
    float *sum3 = S + 768,  *sq3 = S + 1024, *sc3 = S + 1280, *sh3 = S + 1536;
    float *sum4 = S + 1792, *sq4 = S + 2048, *sc4 = S + 2304, *sh4 = S + 2560;
    float *sepc   = S + 2816;                  // BB*256
    float *wvec   = S + 2816 + 16384;          // BB*16*256
    float *cconst = wvec + 262144;             // BB*16
    float *scores = cconst + 1024;             // BB*16*TT
    float *yvec   = scores + BB * 16 * TT;     // BB*256
    // f16 weight copies (zero-padded rows to KPAD), 16B-aligned region.
    _Float16* wf1 = (_Float16*)(yvec + BB * 256);   //  64 x  32
    _Float16* wf2 = wf1 + 64 * 32;                  // 128 x  64
    _Float16* wf3 = wf2 + 128 * 64;                 // 256 x 128
    _Float16* wfi = wf3 + 256 * 128;                // 256 x 256

    const int RT = NN / 16;  // 1460 row tiles

    // Zero BN accumulators (fresh every call — graph replays must match).
    zero_f32_k<<<12, 256, 0, stream>>>(S, 2816);
    // Convert weights to padded f16 once per call.
    wconv_k<<<(64 * 32 + 255) / 256, 256, 0, stream>>>(w1, wf1, 64, 10, 32);
    wconv_k<<<(128 * 64 + 255) / 256, 256, 0, stream>>>(w2, wf2, 128, 64, 64);
    wconv_k<<<(256 * 128 + 255) / 256, 256, 0, stream>>>(w3, wf3, 256, 128, 128);
    wconv_k<<<(256 * 256 + 255) / 256, 256, 0, stream>>>(wi, wfi, 256, 256, 256);

    // Encoder layer 1: x(fp32) [N,10] -> y1 [N,64]
    gemm_bn_k<10, 64, 0><<<RT, 256, 0, stream>>>(x, wf1, b1, nullptr, nullptr,
                                                 bufC, sum1, sq1);
    bn_finalize_k<<<1, 256, 0, stream>>>(sum1, sq1, g1, e1, sc1, sh1, 64, 1.0f / NN);
    // Layer 2: normrelu(y1) -> y2 [N,128]
    gemm_bn_k<64, 128, 1><<<RT, 256, 0, stream>>>(bufC, wf2, b2, sc1, sh1,
                                                  bufB, sum2, sq2);
    bn_finalize_k<<<1, 256, 0, stream>>>(sum2, sq2, g2, e2, sc2, sh2, 128, 1.0f / NN);
    // Layer 3: normrelu(y2) -> y3 [N,256]
    gemm_bn_k<128, 256, 1><<<RT, 256, 0, stream>>>(bufB, wf3, b3, sc2, sh2,
                                                   bufA, sum3, sq3);
    bn_finalize_k<<<1, 256, 0, stream>>>(sum3, sq3, g3, e3, sc3, sh3, 256, 1.0f / NN);

    // SE channel attention
    se_pool_k<<<BB, 256, 0, stream>>>(bufA, sc3, sh3, ws1, ws2, sepc);
    se_apply_k<<<NN, 256, 0, stream>>>(bufA, sc3, sh3, sepc, bufB);  // outp -> bufB

    // inconv: outp -> y4 [N,256]
    gemm_bn_k<256, 256, 2><<<RT, 256, 0, stream>>>(bufB, wfi, bi, nullptr, nullptr,
                                                   bufC, sum4, sq4);
    bn_finalize_k<<<1, 256, 0, stream>>>(sum4, sq4, gi, ei, sc4, sh4, 256, 1.0f / NN);

    // e = normrelu(y4) + positional encoding  -> bufA
    e_build_k<<<NN, 256, 0, stream>>>(bufC, sc4, sh4, positions, bufA);

    // folded attention: kbar/wvec, scores, softmax + V-sum
    kbar_wvec_k<<<BB, 256, 0, stream>>>(bufA, wk, bk, wq, bq, wvec, cconst);
    scores_k<<<(BB * TT * 16 + 255) / 256, 256, 0, stream>>>(bufA, wvec, cconst, scores);
    softmax_y_k<<<BB * 16, 128, 0, stream>>>(scores, bufA, yvec);

    // fused head MLP + decoder + classifier -> logits [64,10]
    head_mlp_k<<<1, 256, 0, stream>>>(yvec, wm, bm, gm, em,
                                      wd1, bd1, gd1, ed1,
                                      wd2, bd2, gd2, ed2,
                                      wc, bc, out);
    (void)in_sizes; (void)n_in; (void)out_size; (void)ws_size;
}
